// MoE_32590211842316
// MI455X (gfx1250) — matrix operand order repaired and checked by reading, not compiled, not run on hardware
//
#include <hip/hip_runtime.h>
#include <math.h>

// MoE forward for MI455X (gfx1250, wave32, WMMA bf16 16x16x32 + async LDS copy).
// Shapes: T=4096 tokens, D=768, F=3072, E=8, top-2, capacity=2048.

#define TOK   4096
#define DDIM  768
#define FDIM  3072
#define NEXP  8
#define NTOPK 2
#define CAP   2048
#define NBUCK 16   // E * TOPK buckets, bucket b = k*8 + e

typedef __bf16 bf16;
typedef __bf16 v4bf  __attribute__((ext_vector_type(4)));
typedef __bf16 v8bf  __attribute__((ext_vector_type(8)));
typedef __bf16 v16bf __attribute__((ext_vector_type(16)));
typedef float  v8f   __attribute__((ext_vector_type(8)));

// Build a 16-element bf16 WMMA fragment from two contiguous 8-element chunks.
__device__ __forceinline__ v16bf make_frag(const bf16* p0, const bf16* p1) {
  v8bf lo = *(const v8bf*)p0;
  v8bf hi = *(const v8bf*)p1;
  v16bf r;
#pragma unroll
  for (int i = 0; i < 8; ++i) { r[i] = lo[i]; r[i + 8] = hi[i]; }
  return r;
}

// ---------------------------------------------------------------- zero init
__global__ void k_zero(float* __restrict__ out, unsigned* __restrict__ accum) {
  size_t i = (size_t)blockIdx.x * 256 + threadIdx.x;
  if (i < (size_t)TOK * DDIM + 1) out[i] = 0.f;
  if (i < 64) accum[i] = 0u;  // nrows[16] + psum[8] + cnt[8] + zsum[1]
}

// ------------------------------------------------- x fp32 -> bf16 (row major)
__global__ void k_xbf(const float* __restrict__ x, bf16* __restrict__ xbf) {
  size_t i = ((size_t)blockIdx.x * 256 + threadIdx.x) * 4;
  float4 v = *(const float4*)(x + i);
  v4bf o;
  o[0] = (bf16)v.x; o[1] = (bf16)v.y; o[2] = (bf16)v.z; o[3] = (bf16)v.w;
  *(v4bf*)(xbf + i) = o;
}

// ------------------------------------------- fp32 -> bf16 transposed repack
// dst[e][c][r] = bf16(src[e][r][c]);  R,C multiples of 32.
__global__ void k_tcvt(const float* __restrict__ src, bf16* __restrict__ dst,
                       int R, int C) {
  __shared__ float tile[32][33];
  const int e = blockIdx.z;
  const int r0 = blockIdx.y * 32, c0 = blockIdx.x * 32;
  const float* s = src + (size_t)e * R * C;
  bf16* d = dst + (size_t)e * R * C;
  const int cx = threadIdx.x & 31;
  const int ry = threadIdx.x >> 5;  // 0..7
#pragma unroll
  for (int i = 0; i < 4; ++i)
    tile[ry + i * 8][cx] = s[(size_t)(r0 + ry + i * 8) * C + c0 + cx];
  __syncthreads();
#pragma unroll
  for (int i = 0; i < 4; ++i)
    d[(size_t)(c0 + ry + i * 8) * R + r0 + cx] = (bf16)tile[cx][ry + i * 8];
}

// ----------------------------------------------------------------- router
__global__ void k_router(const float* __restrict__ x, const float* __restrict__ Wgate,
                         int* __restrict__ ik, float* __restrict__ wtk,
                         float* __restrict__ psum, int* __restrict__ cnt,
                         float* __restrict__ zsum) {
  __shared__ float lWg[DDIM * NEXP];
  __shared__ float s_ps[NEXP];
  __shared__ int   s_cnt[NEXP];
  __shared__ float s_z;
  const int tid = threadIdx.x;
  for (int i = tid; i < DDIM * NEXP; i += 256) lWg[i] = Wgate[i];
  if (tid < NEXP) { s_ps[tid] = 0.f; s_cnt[tid] = 0; }
  if (tid == 0) s_z = 0.f;
  __syncthreads();

  const int t = blockIdx.x * 256 + tid;
  float l[NEXP];
#pragma unroll
  for (int e = 0; e < NEXP; ++e) l[e] = 0.f;
  const float* xr = x + (size_t)t * DDIM;
  for (int d = 0; d < DDIM; ++d) {
    float xv = xr[d];
#pragma unroll
    for (int e = 0; e < NEXP; ++e) l[e] = fmaf(xv, lWg[d * NEXP + e], l[e]);
  }
  float m = l[0];
#pragma unroll
  for (int e = 1; e < NEXP; ++e) m = fmaxf(m, l[e]);
  float p[NEXP]; float s = 0.f;
#pragma unroll
  for (int e = 0; e < NEXP; ++e) { p[e] = expf(l[e] - m); s += p[e]; }
  float inv = 1.f / s;
#pragma unroll
  for (int e = 0; e < NEXP; ++e) p[e] *= inv;
  float lse = logf(s) + m;
  // top-2, ties -> lower index (matches jax.lax.top_k)
  int i0 = 0; float p0 = p[0];
#pragma unroll
  for (int e = 1; e < NEXP; ++e) if (p[e] > p0) { p0 = p[e]; i0 = e; }
  int i1 = -1; float p1 = -1.f;
#pragma unroll
  for (int e = 0; e < NEXP; ++e) if (e != i0 && p[e] > p1) { p1 = p[e]; i1 = e; }
  float wsn = p0 + p1;
  ik[t * 2 + 0] = i0; ik[t * 2 + 1] = i1;
  wtk[t * 2 + 0] = p0 / wsn; wtk[t * 2 + 1] = p1 / wsn;
#pragma unroll
  for (int e = 0; e < NEXP; ++e) atomicAdd(&s_ps[e], p[e]);
  atomicAdd(&s_z, lse * lse);
  atomicAdd(&s_cnt[i0], 1);
  atomicAdd(&s_cnt[i1], 1);
  __syncthreads();
  if (tid < NEXP) { atomicAdd(&psum[tid], s_ps[tid]); atomicAdd(&cnt[tid], s_cnt[tid]); }
  if (tid == 0) atomicAdd(zsum, s_z);
}

// ------------------------------------------- capacity assignment (per e,k)
__global__ void k_assign(const int* __restrict__ ik, const float* __restrict__ wtk,
                         int* __restrict__ rows, float* __restrict__ rwt,
                         int* __restrict__ nrows) {
  const int wave = threadIdx.x >> 5;  // 0..15
  const int lane = threadIdx.x & 31;
  const int k = wave >> 3;
  const int e = wave & 7;
  const int b = k * NEXP + e;
  int cnt = 0;
  for (int base = 0; base < TOK; base += 32) {
    int t = base + lane;
    int sel = (ik[t * 2 + k] == e) ? 1 : 0;
    unsigned mask = (unsigned)__ballot(sel);
    if (sel) {
      int pos = cnt + __popc(mask & ((1u << lane) - 1u));
      if (pos < CAP) {
        rows[b * CAP + pos] = t;
        rwt[b * CAP + pos]  = wtk[t * 2 + k];
      }
    }
    cnt += __popc(mask);
  }
  if (lane == 0) nrows[b] = cnt < CAP ? cnt : CAP;
}

// ----------------------------------------------------------- fused expert FFN
// Block = 8 waves, one 32-token row tile of one (e,k) bucket, full D output.
// Weights pre-packed bf16: WgT/WuT = [E][F][D] (K-contig), WoT = [E][D][F].
__global__ void __launch_bounds__(256, 1)
k_ffn(const bf16* __restrict__ xbf,
      const bf16* __restrict__ WgT, const bf16* __restrict__ WuT,
      const bf16* __restrict__ WoT,
      const int* __restrict__ rows, const float* __restrict__ rwt,
      const int* __restrict__ nrows, float* __restrict__ out) {
  __shared__ bf16  Xs[32][776];  // 32 rows x 768 d (pad 776: conflict-free b128)
  __shared__ bf16  Hs[32][72];   // 32 rows x 64 f (pad 72)
  __shared__ int   ridx[32];
  __shared__ float rw[32];

  const int b = blockIdx.y;
  const int e = b & 7;
  const int n = nrows[b];
  const int row0 = blockIdx.x * 32;
  if (row0 >= n) return;
  const int nval = min(32, n - row0);

  const int tid   = threadIdx.x;
  const int lane  = tid & 31;
  const int wave  = tid >> 5;       // 0..7
  const int lhalf = (lane < 16) ? 0 : 1;
  const int l16   = lane & 15;

  if (tid < 32) {
    if (tid < nval) { ridx[tid] = rows[b * CAP + row0 + tid]; rw[tid] = rwt[b * CAP + row0 + tid]; }
    else            { ridx[tid] = 0; rw[tid] = 0.f; }  // pad rows: token 0, weight 0
  }
  __syncthreads();

  // Async gather of the 32 bf16 token rows straight into LDS
  // (GLOBAL_LOAD_ASYNC_TO_LDS_B128, tracked by ASYNCcnt). Pad rows copy
  // token 0's data; they are masked at output time, so no zero-fill needed.
  {
    const unsigned lds_base = (unsigned)(uintptr_t)&Xs[0][0];
#pragma unroll
    for (int i = 0; i < 12; ++i) {            // 32 rows * 96 16B-chunks / 256 thr
      int c   = tid + i * 256;
      int j   = c / 96, c16 = c % 96;
      const bf16* gp = xbf + (size_t)ridx[j] * DDIM + c16 * 8;
      unsigned lo = lds_base + (unsigned)(j * 776 + c16 * 8) * 2;
      asm volatile("global_load_async_to_lds_b128 %0, %1, off"
                   :: "v"(lo), "v"(gp) : "memory");
    }
    asm volatile("s_wait_asynccnt 0x0" ::: "memory");
  }

  v8f acc[2][6];  // persistent output accumulators: 2 M-tiles x 6 N-tiles
#pragma unroll
  for (int m2 = 0; m2 < 2; ++m2)
#pragma unroll
    for (int nt = 0; nt < 6; ++nt)
#pragma unroll
      for (int r = 0; r < 8; ++r) acc[m2][nt][r] = 0.f;

  const int Mt = wave & 1;    // which 16-row half of the tile
  const int Ft = wave >> 1;   // which 16-wide f sub-chunk of 64
  const size_t wbase_g = (size_t)e * FDIM * DDIM;
  const size_t wbase_o = (size_t)e * DDIM * FDIM;
  const bf16* arow = &Xs[Mt * 16 + l16][lhalf * 16];

  __syncthreads();

  for (int fc = 0; fc < FDIM; fc += 64) {
    const int fcol = fc + Ft * 16 + l16;
    const bf16* bgc = WgT + wbase_g + (size_t)fcol * DDIM + lhalf * 16;
    const bf16* buc = WuT + wbase_g + (size_t)fcol * DDIM + lhalf * 16;
    // Warm L2/WGP$ for the next chunk's weight streams (global_prefetch_b8).
    if (fc + 64 < FDIM) {
      __builtin_prefetch(bgc + (size_t)64 * DDIM, 0, 1);
      __builtin_prefetch(buc + (size_t)64 * DDIM, 0, 1);
      __builtin_prefetch(WoT + wbase_o + (size_t)(wave * 96 + l16) * FDIM + fc + 64, 0, 1);
    }
    // ---- GEMM 1+2: H(32x64) = silu(X@Wg) * (X@Wu), wave computes one 16x16.
    v8f ag, au;
#pragma unroll
    for (int r = 0; r < 8; ++r) { ag[r] = 0.f; au[r] = 0.f; }
    for (int ks = 0; ks < DDIM; ks += 32) {
      v16bf af = make_frag(arow + ks, arow + ks + 8);
      v16bf bg = make_frag(bgc + ks, bgc + ks + 8);
      v16bf bu = make_frag(buc + ks, buc + ks + 8);
      ag = __builtin_amdgcn_wmma_f32_16x16x32_bf16(false, af, false, bg, (short)0, ag, false, false);
      au = __builtin_amdgcn_wmma_f32_16x16x32_bf16(false, af, false, bu, (short)0, au, false, false);
    }
    __syncthreads();  // previous chunk's Hs consumers are done
    {
      const int hr = Mt * 16 + lhalf * 8;
      const int hc = Ft * 16 + l16;
#pragma unroll
      for (int r = 0; r < 8; ++r) {
        float g = ag[r], u = au[r];
        float h = (g / (1.f + expf(-g))) * u;  // silu(g) * u
        Hs[hr + r][hc] = (bf16)h;
      }
    }
    __syncthreads();

    // ---- GEMM 3: Out(32x768) += H(32x64) @ Wo(64x768)
#pragma unroll
    for (int ks2 = 0; ks2 < 2; ++ks2) {
      const int kb = ks2 * 32;
      const bf16* h0 = &Hs[l16][kb + lhalf * 16];
      const bf16* h1 = &Hs[16 + l16][kb + lhalf * 16];
      v16bf a0 = make_frag(h0, h0 + 8);
      v16bf a1 = make_frag(h1, h1 + 8);
#pragma unroll
      for (int nt = 0; nt < 6; ++nt) {
        const int col = wave * 96 + nt * 16 + l16;
        const bf16* bp = WoT + wbase_o + (size_t)col * FDIM + fc + kb + lhalf * 16;
        v16bf bf = make_frag(bp, bp + 8);
        acc[0][nt] = __builtin_amdgcn_wmma_f32_16x16x32_bf16(false, a0, false, bf, (short)0, acc[0][nt], false, false);
        acc[1][nt] = __builtin_amdgcn_wmma_f32_16x16x32_bf16(false, a1, false, bf, (short)0, acc[1][nt], false, false);
      }
    }
  }

  // Weighted scatter-add into the output (two experts may hit the same token).
#pragma unroll
  for (int m2 = 0; m2 < 2; ++m2) {
#pragma unroll
    for (int r = 0; r < 8; ++r) {
      int j = m2 * 16 + lhalf * 8 + r;  // row within tile (C layout: M=r+8*half)
      if (j < nval) {
        float wgt = rw[j];
        float* orow = out + (size_t)ridx[j] * DDIM;
#pragma unroll
        for (int nt = 0; nt < 6; ++nt) {
          int col = wave * 96 + nt * 16 + l16;
          atomicAdd(orow + col, wgt * acc[m2][nt][r]);
        }
      }
    }
  }
}

// -------------------------------------------------------------- aux loss
__global__ void k_aux(const float* __restrict__ psum, const int* __restrict__ cnt,
                      const float* __restrict__ zsum, float* __restrict__ auxout) {
  if (threadIdx.x == 0 && blockIdx.x == 0) {
    float lb = 0.f;
    for (int e = 0; e < NEXP; ++e) {
      float ff = (float)cnt[e] / (float)(TOK * NTOPK);
      float P  = psum[e] / (float)TOK;
      lb += ff * P;
    }
    lb *= (float)NEXP;
    float z = zsum[0] / (float)TOK;
    auxout[0] = 0.01f * lb + 0.001f * z;
  }
}

extern "C" void kernel_launch(void* const* d_in, const int* in_sizes, int n_in,
                              void* d_out, int out_size, void* d_ws, size_t ws_size,
                              hipStream_t stream) {
  const float* x     = (const float*)d_in[0];
  const float* Wgate = (const float*)d_in[1];
  const float* Wg    = (const float*)d_in[2];
  const float* Wu    = (const float*)d_in[3];
  const float* Wo    = (const float*)d_in[4];
  float* out = (float*)d_out;

  char* ws = (char*)d_ws;
  int*   ik    = (int*)  (ws + 0);        // [TOK*2]
  float* wtk   = (float*)(ws + 32768);    // [TOK*2]
  int*   rows  = (int*)  (ws + 65536);    // [16*CAP]
  float* rwt   = (float*)(ws + 196608);   // [16*CAP]
  int*   nrows = (int*)  (ws + 327680);   // [16]
  float* psum  = (float*)(ws + 327744);   // [8]
  int*   cnt   = (int*)  (ws + 327776);   // [8]
  float* zsum  = (float*)(ws + 327808);   // [1]
  bf16*  WgT   = (bf16*) (ws + 327936);   // [E][F][D] bf16
  const size_t wsz = (size_t)NEXP * DDIM * FDIM;
  bf16* WuT = WgT + wsz;                  // [E][F][D] bf16
  bf16* WoT = WuT + wsz;                  // [E][D][F] bf16
  bf16* xbf = WoT + wsz;                  // [TOK][D]  bf16

  k_zero<<<(TOK * DDIM + 256) / 256, 256, 0, stream>>>(out, (unsigned*)nrows);
  k_xbf<<<(TOK * DDIM / 4) / 256, 256, 0, stream>>>(x, xbf);
  k_tcvt<<<dim3(FDIM / 32, DDIM / 32, NEXP), 256, 0, stream>>>(Wg, WgT, DDIM, FDIM);
  k_tcvt<<<dim3(FDIM / 32, DDIM / 32, NEXP), 256, 0, stream>>>(Wu, WuT, DDIM, FDIM);
  k_tcvt<<<dim3(DDIM / 32, FDIM / 32, NEXP), 256, 0, stream>>>(Wo, WoT, FDIM, DDIM);
  k_router<<<TOK / 256, 256, 0, stream>>>(x, Wgate, ik, wtk, psum, cnt, zsum);
  k_assign<<<1, 512, 0, stream>>>(ik, wtk, rows, rwt, nrows);
  k_ffn<<<dim3(CAP / 32, NBUCK), 256, 0, stream>>>(xbf, WgT, WuT, WoT, rows, rwt, nrows, out);
  k_aux<<<1, 32, 0, stream>>>(psum, cnt, zsum, out + (size_t)TOK * DDIM);
}
